// BiSSMBlock_58909771432725
// MI455X (gfx1250) — compile-verified, matrix operand order checked by reading, not run on hardware
//
#include <hip/hip_runtime.h>
#include <hip/hip_bf16.h>
#include <math.h>

// ---------------------------------------------------------------------------
// Bidirectional Mamba block for gfx1250 (MI455X).
// Dense algebra runs through v_wmma_f32_16x16x32_bf16 with a hi/lo bf16
// split of fp32 operands (3 WMMAs per tile k-step) for fp32-class accuracy.
// GEMM uses double-buffered LDS staging with vectorized float4 global loads.
// ---------------------------------------------------------------------------

#define D_MODEL   768
#define D_STATE   16
#define D_CONV    4
#define D_INNER   1536
#define DT_RANK   48
#define BATCH     2
#define SEQLEN    1024
#define M_ROWS    (BATCH * SEQLEN)        // 2048
#define XZ_N      (2 * D_INNER)           // 3072
#define DBL_N     (DT_RANK + 2 * D_STATE) // 80

typedef __attribute__((ext_vector_type(4)))  float          v4f;
typedef __attribute__((ext_vector_type(8)))  float          v8f;
typedef __attribute__((ext_vector_type(16))) __bf16         v16bf;
typedef __attribute__((ext_vector_type(4)))  unsigned short v4u;
typedef __attribute__((ext_vector_type(8)))  unsigned short v8u;
typedef __attribute__((ext_vector_type(16))) unsigned short v16u;

// fp32 -> bf16 round-to-nearest-even (bit manipulation; no __bf16 arithmetic)
__device__ __forceinline__ unsigned short f2bf(float f) {
    unsigned u = __builtin_bit_cast(unsigned, f);
    unsigned r = u + 0x7FFFu + ((u >> 16) & 1u);
    return (unsigned short)(r >> 16);
}
__device__ __forceinline__ float bf2f(unsigned short h) {
    return __builtin_bit_cast(float, (unsigned)h << 16);
}
__device__ __forceinline__ float silu_f(float x) {
    return x / (1.0f + __expf(-x));
}

// Assemble a 16-element bf16 WMMA fragment from two contiguous 16B LDS chunks.
// ISA layout: lanes 0-15 hold K {0..7, 16..23}, lanes 16-31 hold K {8..15,
// 24..31}; caller passes p = rowBase + (lane>=16 ? 8 : 0).
__device__ __forceinline__ v16bf frag_load(const unsigned short* p) {
    v8u lo = *(const v8u*)(p);        // K chunk +0..7
    v8u hi = *(const v8u*)(p + 16);   // K chunk +16..23
    v16u u;
#pragma unroll
    for (int j = 0; j < 8; ++j) { u[j] = lo[j]; u[8 + j] = hi[j]; }
    return __builtin_bit_cast(v16bf, u);
}

// ---------------------------------------------------------------------------
// Generic GEMM: C[:, coff:coff+N] = epilogue(A(MxK, lda) @ W(KxN, row-major))
// Block tile 128x64, 8 waves, each wave a 32x32 region (2x2 WMMA tiles).
// Split-precision bf16: (Ah+Al)(Bh+Bl) ~= AhBh + AhBl + AlBh.
// Double-buffered LDS; float4 global loads (all K/N/lda are multiples of 4).
// mode: 0 = none, 1 = +bias[col], 2 = softplus(x + bias[col])
// ---------------------------------------------------------------------------
#define BM 128
#define BN 64
#define KT 32
#define LSTR 40   // LDS row stride (elements): 80B rows -> 16B aligned, conflict-free

__global__ __launch_bounds__(256) void gemm_bf16x3(
    const float* __restrict__ A, int lda,
    const float* __restrict__ W, // K x N
    float* __restrict__ C, int ldc, int coff,
    int K, int N,
    const float* __restrict__ bias, int mode)
{
    __shared__ unsigned short AsH[2][BM * LSTR];
    __shared__ unsigned short AsL[2][BM * LSTR];
    __shared__ unsigned short BsH[2][BN * LSTR];  // stored transposed: [n][k]
    __shared__ unsigned short BsL[2][BN * LSTR];

    const int tid  = threadIdx.x;
    const int lane = tid & 31;
    const int wave = tid >> 5;
    const int m0   = blockIdx.y * BM;
    const int n0   = blockIdx.x * BN;

    const int wr  = (wave >> 1) * 32;   // wave region row within block tile
    const int wc  = (wave & 1)  * 32;   // wave region col within block tile
    const int kh  = lane >> 4;          // half-wave select
    const int l16 = lane & 15;

    const v4f vzero = {0.f, 0.f, 0.f, 0.f};
    v4f ra[4], rb[2];                   // per-thread prefetch registers

    // fetch one (BM x KT) A tile and (KT x BN) B tile into registers
    auto fetch = [&](int k0) {
#pragma unroll
        for (int i = 0; i < 4; ++i) {           // A: 1024 float4 chunks / 256 thr
            int c = i * 256 + tid;
            int r = c >> 3, k = (c & 7) * 4;    // 8 chunks per 32-wide row
            int gk = k0 + k;
            ra[i] = (gk < K) ? *(const v4f*)(A + (size_t)(m0 + r) * lda + gk)
                             : vzero;
        }
#pragma unroll
        for (int i = 0; i < 2; ++i) {           // B: 512 float4 chunks / 256 thr
            int c = i * 256 + tid;
            int k = c >> 4, n = (c & 15) * 4;   // 16 chunks per 64-wide row
            int gk = k0 + k, gn = n0 + n;
            rb[i] = (gk < K && gn < N) ? *(const v4f*)(W + (size_t)gk * N + gn)
                                       : vzero;
        }
    };

    // convert registers to hi/lo bf16 and store into LDS buffer `buf`
    auto stage = [&](int buf) {
#pragma unroll
        for (int i = 0; i < 4; ++i) {
            int c = i * 256 + tid;
            int r = c >> 3, k = (c & 7) * 4;
            v4u hv, lv;
#pragma unroll
            for (int j = 0; j < 4; ++j) {
                unsigned short h = f2bf(ra[i][j]);
                hv[j] = h;
                lv[j] = f2bf(ra[i][j] - bf2f(h));
            }
            *(v4u*)(&AsH[buf][r * LSTR + k]) = hv;   // ds_store_b64, 8B aligned
            *(v4u*)(&AsL[buf][r * LSTR + k]) = lv;
        }
#pragma unroll
        for (int i = 0; i < 2; ++i) {
            int c = i * 256 + tid;
            int k = c >> 4, n = (c & 15) * 4;
#pragma unroll
            for (int j = 0; j < 4; ++j) {            // transposed: [n][k]
                unsigned short h = f2bf(rb[i][j]);
                BsH[buf][(n + j) * LSTR + k] = h;
                BsL[buf][(n + j) * LSTR + k] = f2bf(rb[i][j] - bf2f(h));
            }
        }
    };

    v8f acc[2][2];
#pragma unroll
    for (int i = 0; i < 2; ++i)
#pragma unroll
        for (int j = 0; j < 2; ++j)
#pragma unroll
            for (int r = 0; r < 8; ++r) acc[i][j][r] = 0.0f;

    // prologue: stage tile 0 into buffer 0
    fetch(0);
    stage(0);

    int p = 0;
    for (int k0 = 0; k0 < K; k0 += KT) {
        __syncthreads();                        // buf[p] visible; buf[p^1] free
        const bool more = (k0 + KT) < K;
        if (more) fetch(k0 + KT);               // global loads overlap WMMAs

        v16bf aH[2], aL[2], bH[2], bL[2];
#pragma unroll
        for (int i = 0; i < 2; ++i) {
            int off = (wr + i * 16 + l16) * LSTR + kh * 8;
            aH[i] = frag_load(&AsH[p][off]);
            aL[i] = frag_load(&AsL[p][off]);
        }
#pragma unroll
        for (int j = 0; j < 2; ++j) {
            int off = (wc + j * 16 + l16) * LSTR + kh * 8;
            bH[j] = frag_load(&BsH[p][off]);
            bL[j] = frag_load(&BsL[p][off]);
        }
#pragma unroll
        for (int i = 0; i < 2; ++i)
#pragma unroll
            for (int j = 0; j < 2; ++j) {
                acc[i][j] = __builtin_amdgcn_wmma_f32_16x16x32_bf16(
                    false, aH[i], false, bH[j], (short)0, acc[i][j], false, false);
                acc[i][j] = __builtin_amdgcn_wmma_f32_16x16x32_bf16(
                    false, aH[i], false, bL[j], (short)0, acc[i][j], false, false);
                acc[i][j] = __builtin_amdgcn_wmma_f32_16x16x32_bf16(
                    false, aL[i], false, bH[j], (short)0, acc[i][j], false, false);
            }

        if (more) stage(p ^ 1);
        p ^= 1;
    }

    // ---- epilogue: D layout VGPR r -> row = base + half*8 + r, col = base + l16
#pragma unroll
    for (int i = 0; i < 2; ++i)
#pragma unroll
        for (int j = 0; j < 2; ++j) {
            int col = n0 + wc + j * 16 + l16;
            if (col < N) {
#pragma unroll
                for (int r = 0; r < 8; ++r) {
                    int row = m0 + wr + i * 16 + kh * 8 + r;
                    float v = acc[i][j][r];
                    if (mode >= 1) v += bias[col];
                    if (mode == 2) v = (v > 20.0f) ? v : log1pf(__expf(v));
                    C[(size_t)row * ldc + coff + col] = v;
                }
            }
        }
}

// ---------------------------------------------------------------------------
// LayerNorm over last dim (768), one block per row
// ---------------------------------------------------------------------------
__global__ __launch_bounds__(256) void ln_kernel(
    const float* __restrict__ x, const float* __restrict__ g,
    const float* __restrict__ b, float* __restrict__ xn)
{
    __shared__ float sb[256], sb2[256];
    const int row = blockIdx.x, tid = threadIdx.x;
    const float* xr = x + (size_t)row * D_MODEL;
    float v[3], s = 0.0f, s2 = 0.0f;
#pragma unroll
    for (int i = 0; i < 3; ++i) {
        v[i] = xr[tid + i * 256];
        s += v[i]; s2 += v[i] * v[i];
    }
    sb[tid] = s; sb2[tid] = s2;
    __syncthreads();
    for (int off = 128; off > 0; off >>= 1) {
        if (tid < off) { sb[tid] += sb[tid + off]; sb2[tid] += sb2[tid + off]; }
        __syncthreads();
    }
    float mu  = sb[0]  * (1.0f / D_MODEL);
    float var = sb2[0] * (1.0f / D_MODEL) - mu * mu;
    float inv = rsqrtf(var + 1e-5f);
#pragma unroll
    for (int i = 0; i < 3; ++i) {
        int c = tid + i * 256;
        xn[(size_t)row * D_MODEL + c] = (v[i] - mu) * inv * g[c] + b[c];
    }
}

// ---------------------------------------------------------------------------
// Causal (or reversed/anti-causal) depthwise conv K=4 + bias + SiLU.
// Reads the xi half (cols 0..1535) of xz (ld = 3072), writes xc (ld = 1536).
// ---------------------------------------------------------------------------
__global__ __launch_bounds__(256) void conv_silu_kernel(
    const float* __restrict__ xz, const float* __restrict__ cw,
    const float* __restrict__ cb, float* __restrict__ xc, int reverse)
{
    int idx = blockIdx.x * 256 + threadIdx.x;
    if (idx >= M_ROWS * D_INNER) return;
    int row = idx / D_INNER, d = idx - row * D_INNER;
    int bt = row >> 10, t = row & 1023;
    float acc = cb[d];
#pragma unroll
    for (int jj = 0; jj < D_CONV; ++jj) {
        int ts = reverse ? (t + jj) : (t - jj);
        if (ts >= 0 && ts < SEQLEN)
            acc += cw[d * D_CONV + (D_CONV - 1 - jj)] *
                   xz[(size_t)(bt * SEQLEN + ts) * XZ_N + d];
    }
    xc[(size_t)row * D_INNER + d] = silu_f(acc);
}

// ---------------------------------------------------------------------------
// Selective scan: one thread per (batch, channel); 16 states in registers.
// Fuses +u*Dp and the SiLU(z) gate. reverse=1 walks the sequence backwards.
// ---------------------------------------------------------------------------
__global__ __launch_bounds__(256) void scan_kernel(
    const float* __restrict__ xc, const float* __restrict__ delta,
    const float* __restrict__ dbl, const float* __restrict__ xz,
    const float* __restrict__ A_log, const float* __restrict__ Dp,
    float* __restrict__ y, int reverse)
{
    int tid = blockIdx.x * 256 + threadIdx.x;
    if (tid >= BATCH * D_INNER) return;
    int bt = tid / D_INNER, d = tid - bt * D_INNER;

    float Av[D_STATE], h[D_STATE];
#pragma unroll
    for (int n = 0; n < D_STATE; ++n) {
        Av[n] = -__expf(A_log[d * D_STATE + n]);
        h[n]  = 0.0f;
    }
    const float Dpd = Dp[d];

    for (int step = 0; step < SEQLEN; ++step) {
        int t   = reverse ? (SEQLEN - 1 - step) : step;
        size_t row = (size_t)(bt * SEQLEN + t);
        float dv = delta[row * D_INNER + d];
        float u  = xc[row * D_INNER + d];
        float du = dv * u;
        const float* dl = dbl + row * DBL_N;
        float yv = 0.0f;
#pragma unroll
        for (int n = 0; n < D_STATE; ++n) {
            float bn = dl[DT_RANK + n];
            float cn = dl[DT_RANK + D_STATE + n];
            h[n] = __expf(dv * Av[n]) * h[n] + du * bn;
            yv  += h[n] * cn;
        }
        float z = xz[row * XZ_N + D_INNER + d];
        y[row * D_INNER + d] = (yv + u * Dpd) * silu_f(z);
    }
}

// ---------------------------------------------------------------------------
// Host launcher
// ---------------------------------------------------------------------------
extern "C" void kernel_launch(void* const* d_in, const int* in_sizes, int n_in,
                              void* d_out, int out_size, void* d_ws, size_t ws_size,
                              hipStream_t stream) {
    const float* x        = (const float*)d_in[0];
    const float* ln_g     = (const float*)d_in[1];
    const float* ln_b     = (const float*)d_in[2];
    const float* W_in_f   = (const float*)d_in[3];
    const float* conv_w_f = (const float*)d_in[4];
    const float* conv_b_f = (const float*)d_in[5];
    const float* W_x_f    = (const float*)d_in[6];
    const float* W_dt_f   = (const float*)d_in[7];
    const float* b_dt_f   = (const float*)d_in[8];
    const float* A_log_f  = (const float*)d_in[9];
    const float* Dp_f     = (const float*)d_in[10];
    const float* W_out_f  = (const float*)d_in[11];
    const float* W_in_b   = (const float*)d_in[12];
    const float* conv_w_b = (const float*)d_in[13];
    const float* conv_b_b = (const float*)d_in[14];
    const float* W_x_b    = (const float*)d_in[15];
    const float* W_dt_b   = (const float*)d_in[16];
    const float* b_dt_b   = (const float*)d_in[17];
    const float* A_log_b  = (const float*)d_in[18];
    const float* Dp_b     = (const float*)d_in[19];
    const float* W_out_b  = (const float*)d_in[20];
    const float* W_c      = (const float*)d_in[21];
    const float* b_c      = (const float*)d_in[22];
    float* out = (float*)d_out;

    // workspace layout (floats)
    float* ws = (float*)d_ws;
    size_t o = 0;
    float* xn   = ws + o; o += (size_t)M_ROWS * D_MODEL;
    float* xzf  = ws + o; o += (size_t)M_ROWS * XZ_N;
    float* xzb  = ws + o; o += (size_t)M_ROWS * XZ_N;
    float* xcf  = ws + o; o += (size_t)M_ROWS * D_INNER;
    float* xcb  = ws + o; o += (size_t)M_ROWS * D_INNER;
    float* dblf = ws + o; o += (size_t)M_ROWS * DBL_N;
    float* dblb = ws + o; o += (size_t)M_ROWS * DBL_N;
    float* delf = ws + o; o += (size_t)M_ROWS * D_INNER;
    float* delb = ws + o; o += (size_t)M_ROWS * D_INNER;
    float* yf   = ws + o; o += (size_t)M_ROWS * D_INNER;
    float* yb   = ws + o; o += (size_t)M_ROWS * D_INNER;
    float* cat  = ws + o; o += (size_t)M_ROWS * (2 * D_MODEL);

    dim3 blk(256);
    auto gemm = [&](const float* A, int lda, const float* W, float* C, int ldc,
                    int coff, int K, int N, const float* bias, int mode) {
        dim3 grid((N + BN - 1) / BN, M_ROWS / BM);
        gemm_bf16x3<<<grid, blk, 0, stream>>>(A, lda, W, C, ldc, coff, K, N, bias, mode);
    };

    // 1) layernorm
    ln_kernel<<<M_ROWS, blk, 0, stream>>>(x, ln_g, ln_b, xn);

    // 2) input projections (both directions; flips are row-permutations, no-op for GEMM)
    gemm(xn, D_MODEL, W_in_f, xzf, XZ_N, 0, D_MODEL, XZ_N, nullptr, 0);
    gemm(xn, D_MODEL, W_in_b, xzb, XZ_N, 0, D_MODEL, XZ_N, nullptr, 0);

    // 3) depthwise conv + SiLU (backward dir runs anti-causally on unflipped data)
    int conv_blocks = (M_ROWS * D_INNER + 255) / 256;
    conv_silu_kernel<<<conv_blocks, blk, 0, stream>>>(xzf, conv_w_f, conv_b_f, xcf, 0);
    conv_silu_kernel<<<conv_blocks, blk, 0, stream>>>(xzb, conv_w_b, conv_b_b, xcb, 1);

    // 4) dt/B/C projection
    gemm(xcf, D_INNER, W_x_f, dblf, DBL_N, 0, D_INNER, DBL_N, nullptr, 0);
    gemm(xcb, D_INNER, W_x_b, dblb, DBL_N, 0, D_INNER, DBL_N, nullptr, 0);

    // 5) delta = softplus(dt @ W_dt + b_dt)   (A = dbl[:, :48] via lda=80)
    gemm(dblf, DBL_N, W_dt_f, delf, D_INNER, 0, DT_RANK, D_INNER, b_dt_f, 2);
    gemm(dblb, DBL_N, W_dt_b, delb, D_INNER, 0, DT_RANK, D_INNER, b_dt_b, 2);

    // 6) selective scan (+Dp, +SiLU(z) gate fused)
    int scan_blocks = (BATCH * D_INNER + 255) / 256;
    scan_kernel<<<scan_blocks, blk, 0, stream>>>(xcf, delf, dblf, xzf, A_log_f, Dp_f, yf, 0);
    scan_kernel<<<scan_blocks, blk, 0, stream>>>(xcb, delb, dblb, xzb, A_log_b, Dp_b, yb, 1);

    // 7) output projections into concatenated buffer
    gemm(yf, D_INNER, W_out_f, cat, 2 * D_MODEL, 0,       D_INNER, D_MODEL, nullptr, 0);
    gemm(yb, D_INNER, W_out_b, cat, 2 * D_MODEL, D_MODEL, D_INNER, D_MODEL, nullptr, 0);

    // 8) final fuse: out = cat @ W_c + b_c
    gemm(cat, 2 * D_MODEL, W_c, out, D_MODEL, 0, 2 * D_MODEL, D_MODEL, b_c, 1);
}
